// IRCNN_LSTMCell_34875134443722
// MI455X (gfx1250) — matrix operand order repaired and checked by reading, not compiled, not run on hardware
//
#include <hip/hip_runtime.h>
#include <hip/hip_bf16.h>
#include <stdint.h>

typedef __attribute__((ext_vector_type(16))) __bf16 v16bf;
typedef __attribute__((ext_vector_type(8)))  float  v8f;

__device__ __forceinline__ unsigned short f2bf(float f) {
  unsigned int u = __float_as_uint(f);
  u += 0x7FFFu + ((u >> 16) & 1u);          // round-to-nearest-even
  return (unsigned short)(u >> 16);
}
__device__ __forceinline__ float sigmoidf_(float x) {
  return 1.0f / (1.0f + __expf(-x));
}

// CDNA5 async DMA: global -> LDS, 16 bytes per lane, tracked by ASYNCcnt.
// GV mode: first operand = LDS byte address VGPR, second = 64-bit global address.
__device__ __forceinline__ void async_ld16(unsigned lds_byte, const void* gaddr) {
  asm volatile("global_load_async_to_lds_b128 %0, %1, off"
               :: "v"(lds_byte), "v"(gaddr) : "memory");
}
__device__ __forceinline__ void wait_async0() {
#if __has_builtin(__builtin_amdgcn_s_wait_asynccnt)
  __builtin_amdgcn_s_wait_asynccnt(0);
#else
  asm volatile("s_wait_asynccnt 0" ::: "memory");
#endif
}

// ---------------- pack kernels (f32 -> bf16 workspace) ----------------

// A_cat[4096 x 6144] = [x | hx | cx]
__global__ __launch_bounds__(256) void pack_A_kernel(
    const float* __restrict__ x, const float* __restrict__ hx,
    const float* __restrict__ cx, unsigned short* __restrict__ Acat) {
  const int col = blockIdx.x * 256 + threadIdx.x;   // 0..6143
  const int row = blockIdx.y;                       // 0..4095
  float v;
  if (col < 2048)       v = x [(size_t)row * 2048 + col];
  else if (col < 4096)  v = hx[(size_t)row * 2048 + (col - 2048)];
  else                  v = cx[(size_t)row * 2048 + (col - 4096)];
  Acat[(size_t)row * 6144 + col] = f2bf(v);
}

// W_cat[8192 x 6144]: per output row r (gate = r>>11):
//   cols 0:2048 = w_ih[r], 2048:4096 = w_hh[r],
//   4096:6144 = w_pi (gate 0) / w_pf (gate 1) / zero (gates 2,3: never read, K=4096)
__global__ __launch_bounds__(256) void pack_W_kernel(
    const float* __restrict__ w_ih, const float* __restrict__ w_hh,
    const float* __restrict__ w_pi, const float* __restrict__ w_pf,
    unsigned short* __restrict__ Wcat) {
  const int col = blockIdx.x * 256 + threadIdx.x;   // 0..6143
  const int row = blockIdx.y;                       // 0..8191
  const int gate = row >> 11;
  const int rr = row & 2047;
  float v;
  if (col < 2048)       v = w_ih[(size_t)row * 2048 + col];
  else if (col < 4096)  v = w_hh[(size_t)row * 2048 + (col - 2048)];
  else {
    const int cc = col - 4096;
    if (gate == 0)      v = w_pi[(size_t)rr * 2048 + cc];
    else if (gate == 1) v = w_pf[(size_t)rr * 2048 + cc];
    else                v = 0.0f;
  }
  Wcat[(size_t)row * 6144 + col] = f2bf(v);
}

__global__ __launch_bounds__(256) void pack_wpo_kernel(
    const float* __restrict__ w_po, unsigned short* __restrict__ out) {
  const size_t idx = (size_t)blockIdx.x * 256 + threadIdx.x;
  out[idx] = f2bf(w_po[idx]);
}

// ---------------- WMMA bf16 GEMM: C = A @ W^T ----------------
// Block tile 128(M) x 128(N), staged K = 64 (two 32-wide WMMA sub-steps),
// double-buffered LDS filled by GLOBAL_LOAD_ASYNC_TO_LDS_B128 (no data VGPRs,
// no spills), one s_wait_asynccnt + barrier per staged tile (16 wmma/wave).
// 8 waves as 2x4, each computing 64x32 (8 accumulators).
// EPI==0: gates GEMM (N=8192, per-gate K, epilogue adds b_ih+b_hh, ldc=8192)
// EPI==1: p_o GEMM (N=2048, K=2048, epilogue hy = sigmoid(o_pre + v)*tanh(cy))

#define BK   64
#define LDSP 72   // padded row stride in halfwords (64 data + 8 pad), 144 B = 9*16

template <int EPI>
__global__ __launch_bounds__(256) void gemm_bf16_wmma(
    const unsigned short* __restrict__ A, int lda,
    const unsigned short* __restrict__ W, int ldw,
    float* __restrict__ out,
    const float* __restrict__ e0,   // EPI0: b_ih   EPI1: gates + 6144 (o_pre, ld 8192)
    const float* __restrict__ e1) { // EPI0: b_hh   EPI1: cy f32 (ld 2048)
  __shared__ unsigned short sA[2][128 * LDSP];   // 36 KB
  __shared__ unsigned short sW[2][128 * LDSP];   // 36 KB

  const int t     = threadIdx.x;
  const int lane  = t & 31;
  const int wave  = t >> 5;
  const int waveM = wave >> 2;       // 0..1 -> M offset 64*waveM
  const int waveN = wave & 3;        // 0..3 -> N offset 32*waveN
  const int blockRow = blockIdx.y * 128;
  const int blockCol = blockIdx.x * 128;

  // per-gate K: blocks 0..31 cover i/f gates (cx peephole folded in, K=6144)
  const int K    = (EPI == 0) ? ((blockIdx.x < 32) ? 6144 : 4096) : 2048;
  const int numK = K >> 6;           // 64 per staged tile

  const int chunk = t & 7;           // 8 x 16B chunks per 64-halfword row
  const int row0  = t >> 3;          // 0..31

  // persistent global pointers, bumped by 64 halfwords per staged tile
  const unsigned short* pA = A + (size_t)(blockRow + row0) * lda + chunk * 8;
  const unsigned short* pW = W + (size_t)(blockCol + row0) * ldw + chunk * 8;
  const size_t a32 = (size_t)32 * lda;
  const size_t w32 = (size_t)32 * ldw;

  const int soff = row0 * LDSP + chunk * 8;
  const unsigned baseA[2] = {(unsigned)(uintptr_t)&sA[0][soff],
                             (unsigned)(uintptr_t)&sA[1][soff]};
  const unsigned baseW[2] = {(unsigned)(uintptr_t)&sW[0][soff],
                             (unsigned)(uintptr_t)&sW[1][soff]};

  // issue one staged tile (8 async b128 per thread), directly into LDS buffer p
  auto issue_tile = [&](int p) {
#pragma unroll
    for (int i = 0; i < 4; ++i)
      async_ld16(baseA[p] + i * (32 * LDSP * 2), pA + (size_t)i * a32);
#pragma unroll
    for (int i = 0; i < 4; ++i)
      async_ld16(baseW[p] + i * (32 * LDSP * 2), pW + (size_t)i * w32);
    pA += BK; pW += BK;
  };

  v8f acc[4][2];
  const v8f vzero = {0.f, 0.f, 0.f, 0.f, 0.f, 0.f, 0.f, 0.f};
#pragma unroll
  for (int im = 0; im < 4; ++im)
#pragma unroll
    for (int in = 0; in < 2; ++in) acc[im][in] = vzero;

  const int nl  = lane & 15;
  const int hkA = (lane >> 4) * 8;    // A frag: K chunks at {hkA, hkA+16} halfwords
  const int hkB = (lane >> 4) * 8;    // B frag: K chunks at {hkB, hkB+16}
  union FU { v16bf v; uint4 q[2]; };

  // prologue: tile 0 -> buf 0
  issue_tile(0);
  wait_async0();
  __syncthreads();

  for (int kt = 0; kt < numK; ++kt) {
    const int p = kt & 1;
    if (kt + 1 < numK) issue_tile(p ^ 1);   // async DMA overlaps wmmas below

#pragma unroll
    for (int ks = 0; ks < 2; ++ks) {   // two 32-wide K sub-steps
      const int kb = ks * 32;
      FU fb[2];
#pragma unroll
      for (int in = 0; in < 2; ++in) {
        const int r = waveN * 32 + in * 16 + nl;
        fb[in].q[0] = *(const uint4*)&sW[p][r * LDSP + kb + hkB];
        fb[in].q[1] = *(const uint4*)&sW[p][r * LDSP + kb + hkB + 16];
      }
#pragma unroll
      for (int im = 0; im < 4; ++im) {
        const int r = waveM * 64 + im * 16 + nl;
        FU fa;
        fa.q[0] = *(const uint4*)&sA[p][r * LDSP + kb + hkA];
        fa.q[1] = *(const uint4*)&sA[p][r * LDSP + kb + hkA + 16];
#pragma unroll
        for (int in = 0; in < 2; ++in)
          acc[im][in] = __builtin_amdgcn_wmma_f32_16x16x32_bf16(
              false, fa.v, false, fb[in].v, (short)0, acc[im][in], false, false);
      }
    }

    if (kt + 1 < numK) {
      wait_async0();      // own async copies for tile kt+1 landed
      __syncthreads();    // all waves' copies landed; all reads of buf p done
    }
  }

  // epilogue: C/D layout — VGPR r, lanes 0-15: (M=r, N=lane); lanes 16-31: (M=r+8)
  const int mhalf = (lane >> 4) << 3;
#pragma unroll
  for (int im = 0; im < 4; ++im) {
#pragma unroll
    for (int in = 0; in < 2; ++in) {
      const int col = blockCol + waveN * 32 + in * 16 + nl;
      float bias = 0.0f;
      if (EPI == 0) bias = e0[col] + e1[col];
#pragma unroll
      for (int r = 0; r < 8; ++r) {
        const int row = blockRow + waveM * 64 + im * 16 + mhalf + r;
        const float v = acc[im][in][r];
        if (EPI == 0) {
          out[(size_t)row * 8192 + col] = v + bias;
        } else {
          const float op  = e0[(size_t)row * 8192 + col];   // o-gate preact
          const float og  = sigmoidf_(op + v);
          const float cyv = e1[(size_t)row * 2048 + col];
          out[(size_t)row * 2048 + col] = og * tanhf(cyv);  // hy
        }
      }
    }
  }
}

// ---------------- elementwise cy ----------------
__global__ __launch_bounds__(256) void cy_kernel(
    const float* __restrict__ gates, const float* __restrict__ cx,
    const float* __restrict__ td0, const float* __restrict__ td1,
    const float* __restrict__ a, const float* __restrict__ b,
    float* __restrict__ cy_out, unsigned short* __restrict__ cy_bf) {
  const size_t idx = (size_t)blockIdx.x * 256 + threadIdx.x;
  const int row = (int)(idx >> 11);
  const int col = (int)(idx & 2047);
  const float* g = gates + (size_t)row * 8192;
  const float ip = g[col];
  const float fp = g[col + 2048];
  const float gp = g[col + 4096];
  const float av = a[0], bv = b[0];
  const float forc = av * __expf(-bv * td0[row]);
  const float inc  = av * __expf(-bv * td1[row]);
  const float cyv = forc * sigmoidf_(fp) * cx[idx] + inc * sigmoidf_(ip) * tanhf(gp);
  cy_out[idx] = cyv;
  cy_bf[idx]  = f2bf(cyv);
}

// ---------------- host ----------------
extern "C" void kernel_launch(void* const* d_in, const int* in_sizes, int n_in,
                              void* d_out, int out_size, void* d_ws, size_t ws_size,
                              hipStream_t stream) {
  (void)in_sizes; (void)n_in; (void)out_size; (void)ws_size;
  const float* x    = (const float*)d_in[0];
  const float* td0  = (const float*)d_in[1];
  const float* td1  = (const float*)d_in[2];
  const float* hx   = (const float*)d_in[3];
  const float* cx   = (const float*)d_in[4];
  const float* w_ih = (const float*)d_in[5];
  const float* w_hh = (const float*)d_in[6];
  const float* w_pi = (const float*)d_in[7];
  const float* w_pf = (const float*)d_in[8];
  const float* w_po = (const float*)d_in[9];
  const float* b_ih = (const float*)d_in[10];
  const float* b_hh = (const float*)d_in[11];
  const float* a    = (const float*)d_in[12];
  const float* b    = (const float*)d_in[13];

  // workspace layout (bytes)
  char* ws = (char*)d_ws;
  unsigned short* A_cat = (unsigned short*)(ws);                    //  50,331,648
  unsigned short* W_cat = (unsigned short*)(ws + 50331648ull);      // 100,663,296
  float*          gates = (float*)        (ws + 150994944ull);      // 134,217,728
  unsigned short* wpo   = (unsigned short*)(ws + 285212672ull);     //   8,388,608
  unsigned short* cybf  = (unsigned short*)(ws + 293601280ull);     //  16,777,216
                                                                    // total ~296 MB
  float* hy  = (float*)d_out;
  float* cyf = hy + (size_t)4096 * 2048;   // outputs: [hy | cy]

  pack_A_kernel<<<dim3(24, 4096), 256, 0, stream>>>(x, hx, cx, A_cat);
  pack_W_kernel<<<dim3(24, 8192), 256, 0, stream>>>(w_ih, w_hh, w_pi, w_pf, W_cat);
  pack_wpo_kernel<<<16384, 256, 0, stream>>>(w_po, wpo);

  // gates[4096 x 8192] = A_cat @ W_cat^T (+ b_ih + b_hh), per-gate K
  gemm_bf16_wmma<0><<<dim3(64, 32), 256, 0, stream>>>(
      A_cat, 6144, W_cat, 6144, gates, b_ih, b_hh);

  // cy (f32 to output, bf16 for next GEMM)
  cy_kernel<<<32768, 256, 0, stream>>>(gates, cx, td0, td1, a, b, cyf, cybf);

  // hy = sigmoid(o_pre + cy@w_po^T) * tanh(cy)
  gemm_bf16_wmma<1><<<dim3(8, 32), 256, 0, stream>>>(
      cybf, 2048, wpo, 2048, hy, gates + 6144, cyf);
}
